// GemmRS_15496242004357
// MI455X (gfx1250) — matrix operand order repaired and checked by reading, not compiled, not run on hardware
//
#include <hip/hip_runtime.h>
#include <hip/hip_bf16.h>

// ---------------------------------------------------------------------------
// Fused partial-GEMM + reduce-scatter for MI455X (gfx1250, wave32, WMMA).
//   out[d, m, n] = sum_w sum_k input[w, d*1024+m, k] * weight[w, n, k]
// fp32 in HBM -> bf16 in LDS -> v_wmma_f32_16x16x32_bf16 -> fp32 out.
// Block tile 128x256, 8 waves, wave tile 64x64 (4x4 WMMA fragments).
// ---------------------------------------------------------------------------

typedef __attribute__((ext_vector_type(16))) __bf16 v16bf;
typedef __attribute__((ext_vector_type(8)))  __bf16 v8bf;
typedef __attribute__((ext_vector_type(4)))  __bf16 v4bf;
typedef __attribute__((ext_vector_type(8)))  float  v8f;
typedef __attribute__((ext_vector_type(4)))  float  v4f;

#define WSZ   8       // world size (ranks)
#define MT    8192    // total M rows in input
#define KK    512     // local K
#define NN    1024    // N
#define MS    1024    // rows per dst rank (MT / WSZ)
#define BM    128     // block tile M
#define BN    256     // block tile N
#define BK    32      // K chunk (one bf16 WMMA depth)
#define LDSS  40      // padded LDS row stride (bf16 elems) = 80 B
#define KCH   (KK / BK)        // 16 chunks per rank
#define ITERS (WSZ * KCH)      // 128 total K iterations

__global__ __launch_bounds__(256)
void gemm_rs_wmma(const float* __restrict__ A,   // [WSZ, MT, KK]
                  const float* __restrict__ B,   // [WSZ, NN, KK]
                  float* __restrict__ C)         // [WSZ, MS, NN]
{
    __shared__ __attribute__((aligned(16))) __bf16 sA[2][BM * LDSS];
    __shared__ __attribute__((aligned(16))) __bf16 sB[2][BN * LDSS];

    const int tid  = threadIdx.x;
    const int lane = tid & 31;
    const int wid  = tid >> 5;      // 0..7
    const int wm   = wid >> 2;      // 0..1  (wave row: 64-row slabs)
    const int wn   = wid & 3;       // 0..3  (wave col: 64-col slabs)

    const int d     = blockIdx.z;        // dst rank
    const int mBase = blockIdx.y * BM;   // within Ms
    const int nBase = blockIdx.x * BN;

    // Staging coords: each chunk-stage covers rows in strides of 32, with
    // 8 float4 per 32-float row. A tile: 128x32 (4 rounds). B: 256x32 (8).
    const int sRow = tid >> 3;            // 0..31
    const int sC4  = (tid & 7) * 4;       // k offset within chunk, 0..28

    v8f acc[4][4] = {};

    // ---- stage iteration 0 into buffer 0 -------------------------------
    {
        const float* ap = A + ((size_t)(d * MS + mBase)) * KK + sC4;
        const float* bp = B + ((size_t)nBase) * KK + sC4;
#pragma unroll
        for (int j = 0; j < 4; ++j) {
            v4f av = *(const v4f*)(ap + (size_t)(sRow + j * 32) * KK);
            *(v4bf*)&sA[0][(sRow + j * 32) * LDSS + sC4] = __builtin_convertvector(av, v4bf);
        }
#pragma unroll
        for (int j = 0; j < 8; ++j) {
            v4f bv = *(const v4f*)(bp + (size_t)(sRow + j * 32) * KK);
            *(v4bf*)&sB[0][(sRow + j * 32) * LDSS + sC4] = __builtin_convertvector(bv, v4bf);
        }
    }
    __syncthreads();

    // Fragment addressing (ISA 7.12.2):
    // A frag (16x32 MxK): lane holds row M = lane%16; lanes<16 get K {0..7,16..23},
    // lanes>=16 get K {8..15,24..31} -> two 16B LDS reads at +0/+16 elems from
    // K base (lane&16)?8:0.
    // B frag (32x16 KxN): lane holds col N = lane%16; same K split as A
    // -> two 16B reads at +0/+16 elems from K base (lane&16)?8:0.
    const int aRow = wm * 64 + (lane & 15);
    const int aK   = (lane & 16) ? 8 : 0;
    const int bRow = wn * 64 + (lane & 15);
    const int bK   = (lane & 16) ? 8 : 0;

    for (int it = 0; it < ITERS; ++it) {
        const int cur = it & 1;

        // ---- prefetch next tile (global -> regs) -----------------------
        v4f avr[4], bvr[8];
        const bool hasNext = (it + 1 < ITERS);
        if (hasNext) {
            const int nit = it + 1;
            const int w   = nit >> 4;          // rank index
            const int kb  = (nit & 15) * BK;   // k base within rank
            const float* ap = A + ((size_t)w * MT + (size_t)(d * MS + mBase)) * KK + kb + sC4;
            const float* bp = B + ((size_t)w * NN + (size_t)nBase) * KK + kb + sC4;
#pragma unroll
            for (int j = 0; j < 4; ++j)
                avr[j] = *(const v4f*)(ap + (size_t)(sRow + j * 32) * KK);
#pragma unroll
            for (int j = 0; j < 8; ++j)
                bvr[j] = *(const v4f*)(bp + (size_t)(sRow + j * 32) * KK);
        }

        // ---- load fragments from LDS -----------------------------------
        union frag { v16bf v; v8bf h[2]; };
        frag afr[4], bfr[4];
        const __bf16* a0 = sA[cur];
        const __bf16* b0 = sB[cur];
#pragma unroll
        for (int i = 0; i < 4; ++i) {
            const __bf16* p = a0 + (aRow + i * 16) * LDSS + aK;
            afr[i].h[0] = *(const v8bf*)(p);
            afr[i].h[1] = *(const v8bf*)(p + 16);
        }
#pragma unroll
        for (int j = 0; j < 4; ++j) {
            const __bf16* p = b0 + (bRow + j * 16) * LDSS + bK;
            bfr[j].h[0] = *(const v8bf*)(p);
            bfr[j].h[1] = *(const v8bf*)(p + 16);
        }

        // ---- 16 WMMA ops: 64x64 wave tile, K += 32 ---------------------
#pragma unroll
        for (int i = 0; i < 4; ++i)
#pragma unroll
            for (int j = 0; j < 4; ++j)
                acc[i][j] = __builtin_amdgcn_wmma_f32_16x16x32_bf16(
                    /*neg_a=*/false, afr[i].v,
                    /*neg_b=*/false, bfr[j].v,
                    /*c_mod=*/(short)0, acc[i][j],
                    /*reuse_a=*/false, /*reuse_b=*/false);

        // ---- convert + store prefetched tile into other buffer ---------
        if (hasNext) {
            const int nb = cur ^ 1;
#pragma unroll
            for (int j = 0; j < 4; ++j)
                *(v4bf*)&sA[nb][(sRow + j * 32) * LDSS + sC4] = __builtin_convertvector(avr[j], v4bf);
#pragma unroll
            for (int j = 0; j < 8; ++j)
                *(v4bf*)&sB[nb][(sRow + j * 32) * LDSS + sC4] = __builtin_convertvector(bvr[j], v4bf);
        }
        __syncthreads();
    }

    // ---- write out fp32 C ----------------------------------------------
    // C/D layout: VGPR v, lanes 0-15 -> M = v, N = lane%16; lanes 16-31 -> M = v+8.
    float* outp = C + (size_t)d * MS * NN;
    const int col0 = nBase + wn * 64 + (lane & 15);
    const int row0 = mBase + wm * 64 + ((lane & 16) ? 8 : 0);
#pragma unroll
    for (int i = 0; i < 4; ++i) {
#pragma unroll
        for (int j = 0; j < 4; ++j) {
            const int c  = col0 + j * 16;
            const int r0 = row0 + i * 16;
#pragma unroll
            for (int v = 0; v < 8; ++v)
                outp[(size_t)(r0 + v) * NN + c] = acc[i][j][v];
        }
    }
}

extern "C" void kernel_launch(void* const* d_in, const int* in_sizes, int n_in,
                              void* d_out, int out_size, void* d_ws, size_t ws_size,
                              hipStream_t stream) {
    const float* inp = (const float*)d_in[0];   // [8, 8192, 512] fp32
    const float* wgt = (const float*)d_in[1];   // [8, 1024, 512] fp32
    float* out = (float*)d_out;                 // [8, 1024, 1024] fp32

    dim3 grid(NN / BN, MS / BM, WSZ);           // (4, 8, 8)
    gemm_rs_wmma<<<grid, 256, 0, stream>>>(inp, wgt, out);
}
